// Wide_ResNet_21131239097112
// MI455X (gfx1250) — compile-verified, matrix operand order checked by reading, not run on hardware
//
#include <hip/hip_runtime.h>
#include <hip/hip_bf16.h>

typedef __attribute__((ext_vector_type(16))) _Float16 v16h;
typedef __attribute__((ext_vector_type(8)))  float    v8f;
typedef unsigned int u32x4 __attribute__((ext_vector_type(4)));
typedef int          i32x8 __attribute__((ext_vector_type(8)));
typedef int          i32x4 __attribute__((ext_vector_type(4)));

#define COT 4  // output channels per thread in adder2d

// ---------------------------------------------------------------------------
// TDM: DMA a contiguous 1-D tile of `elems` f32 from global to LDS.
// D# layout per CDNA5 ISA (groups 0/1; groups 2/3 zero).
// Wave-level op (EXEC ignored); tracked by TENSORcnt.
// This toolchain exposes the 6-arg builtin:
//   (u32x4 g0, i32x8 g1, i32x4 g2, i32x4 g3, i32x8 extra, i32 cpol)
// ---------------------------------------------------------------------------
__device__ __forceinline__ void tdm_load_f32_1d(const float* gsrc,
                                                float* lds_dst, int elems)
{
#if __has_builtin(__builtin_amdgcn_tensor_load_to_lds)
    unsigned long long ga = (unsigned long long)(const void*)gsrc;
    // Flat LDS addresses carry the byte offset in addr[31:0] (aperture rule).
    unsigned lds_off = (unsigned)(unsigned long long)(void*)lds_dst;

    u32x4 g0 = {0u, 0u, 0u, 0u};
    g0[0] = 1u;                                       // count=1 (valid user D#)
    g0[1] = lds_off;                                  // lds_addr
    g0[2] = (unsigned)(ga & 0xFFFFFFFFu);             // global_addr[31:0]
    g0[3] = (unsigned)((ga >> 32) & 0x1FFFFFFu)       // global_addr[56:32]
          | (2u << 30);                               // type=2 ("image")

    unsigned ue = (unsigned)elems;
    i32x8 g1 = {0, 0, 0, 0, 0, 0, 0, 0};
    g1[0] = (int)(2u << 16);                          // data_size=2 -> 4 bytes
    g1[1] = (int)((ue & 0xFFFFu) << 16);              // tensor_dim0[15:0]
    g1[2] = (int)(((ue >> 16) & 0xFFFFu) | (1u << 16)); // dim0[31:16], tensor_dim1=1
    g1[3] = (int)((ue & 0xFFFFu) << 16);              // tile_dim0 (elems < 65536)
    g1[5] = (int)ue;                                  // tensor_dim0_stride lo32

    i32x4 z4 = {0, 0, 0, 0};
    i32x8 z8 = {0, 0, 0, 0, 0, 0, 0, 0};
    __builtin_amdgcn_tensor_load_to_lds(g0, g1, z4, z4, z8, 0);
#else
    (void)gsrc; (void)lds_dst; (void)elems;
#endif
}

__device__ __forceinline__ void wait_tensorcnt0()
{
#if __has_builtin(__builtin_amdgcn_s_wait_tensorcnt)
    __builtin_amdgcn_s_wait_tensorcnt(0);
#else
    asm volatile("s_wait_tensorcnt 0x0" ::: "memory");
#endif
}

// ---------------------------------------------------------------------------
// conv1 as implicit GEMM via WMMA f16 (M=16384 positions, N=16 ch, K=27->32)
// Block = 256 threads = 8 waves; each wave computes a 16x16 output tile.
// ---------------------------------------------------------------------------
__global__ void conv1_wmma(const float* __restrict__ x,   // (16,3,32,32)
                           const float* __restrict__ w,   // (16,3,3,3)
                           const float* __restrict__ bias,// (16)
                           float* __restrict__ out)       // (16,16,32,32)
{
    __shared__ _Float16 Bs[32][16]; // [k][n] weight matrix, k>=27 zero-padded
    const int tid = threadIdx.x;
    for (int i = tid; i < 32 * 16; i += 256) {
        int k = i >> 4, n = i & 15;
        _Float16 v = (_Float16)0.f;
        if (k < 27) {
            int c = k / 9, t = k % 9;
            v = (_Float16)w[((n * 3 + c) * 3 + (t / 3)) * 3 + (t % 3)];
        }
        Bs[k][n] = v;
    }
    __syncthreads();

    const int wave = tid >> 5;
    const int lane = tid & 31;
    const int half = lane >> 4;
    const int m    = lane & 15;
    const int tileBase = (blockIdx.x * 8 + wave) * 16;

    const int pos  = tileBase + m;
    const int nimg = pos >> 10;
    const int hw   = pos & 1023;
    const int h    = hw >> 5;
    const int wpix = hw & 31;

    v16h a, bfrag;
#pragma unroll
    for (int i = 0; i < 8; ++i) {
        const int kbase = ((i >= 4) ? 16 : 0) + half * 8 + 2 * (i & 3);
#pragma unroll
        for (int j = 0; j < 2; ++j) {
            const int k = kbase + j;
            float av = 0.f;
            if (k < 27) {
                int c = k / 9, t = k % 9, r = t / 3, s = t % 3;
                int hh = h + r - 1, ww = wpix + s - 1;
                if (hh >= 0 && hh < 32 && ww >= 0 && ww < 32)
                    av = x[((nimg * 3 + c) * 32 + hh) * 32 + ww];
            }
            a[2 * i + j]     = (_Float16)av;
            bfrag[2 * i + j] = Bs[k][lane & 15];
        }
    }

    v8f c = {};
    c = __builtin_amdgcn_wmma_f32_16x16x32_f16(false, a, false, bfrag,
                                               (short)0, c, false, false);

    const int N = lane & 15;
#pragma unroll
    for (int j = 0; j < 8; ++j) {
        const int M   = half * 8 + j;
        const int p   = tileBase + M;
        const int ni  = p >> 10;
        const int phw = p & 1023;
        out[((ni * 16 + N) << 10) + phw] = c[j] + bias[N];
    }
}

// ---------------------------------------------------------------------------
// FC head: (16 x 256) @ (256 x 10, padded to 16) via 8 chained WMMA k-steps.
// Single wave.
// ---------------------------------------------------------------------------
__global__ void fc_wmma(const float* __restrict__ pooled, // (16,256)
                        const float* __restrict__ fw,     // (256,10)
                        const float* __restrict__ fb,     // (10)
                        float* __restrict__ out)          // (16,10)
{
    const int lane = threadIdx.x & 31;
    const int half = lane >> 4;
    const int mn   = lane & 15;

    v8f acc = {};
    for (int k0 = 0; k0 < 256; k0 += 32) {
        v16h a, bfrag;
#pragma unroll
        for (int i = 0; i < 8; ++i) {
            const int kbase = ((i >= 4) ? 16 : 0) + half * 8 + 2 * (i & 3);
#pragma unroll
            for (int j = 0; j < 2; ++j) {
                const int k = k0 + kbase + j;
                a[2 * i + j]     = (_Float16)pooled[mn * 256 + k];
                bfrag[2 * i + j] = (_Float16)((mn < 10) ? fw[k * 10 + mn] : 0.f);
            }
        }
        acc = __builtin_amdgcn_wmma_f32_16x16x32_f16(false, a, false, bfrag,
                                                     (short)0, acc, false, false);
    }
    const int N = lane & 15;
    if (N < 10) {
#pragma unroll
        for (int j = 0; j < 8; ++j) {
            const int M = half * 8 + j;
            out[M * 10 + N] = acc[j] + fb[N];
        }
    }
}

// ---------------------------------------------------------------------------
// Per-channel batchnorm statistics: stats[c]=mean, stats[C+c]=1/sqrt(var+eps)
// ---------------------------------------------------------------------------
__global__ void bn_stats(const float* __restrict__ x, float* __restrict__ stats,
                         int C, int HW, int N)
{
    const int c   = blockIdx.x;
    const int tid = threadIdx.x;
    const int total = N * HW;
    float s = 0.f, s2 = 0.f;
    for (int i = tid; i < total; i += 256) {
        int n = i / HW, r = i - n * HW;
        float v = x[((size_t)n * C + c) * HW + r];
        s += v; s2 += v * v;
    }
    __shared__ float sh[256], sh2[256];
    sh[tid] = s; sh2[tid] = s2;
    __syncthreads();
    for (int off = 128; off > 0; off >>= 1) {
        if (tid < off) { sh[tid] += sh[tid + off]; sh2[tid] += sh2[tid + off]; }
        __syncthreads();
    }
    if (tid == 0) {
        float inv  = 1.0f / (float)total;
        float mean = sh[0] * inv;
        float var  = sh2[0] * inv - mean * mean;
        stats[c]     = mean;
        stats[C + c] = rsqrtf(var + 1e-5f);
    }
}

// ---------------------------------------------------------------------------
// Fused BN + ReLU elementwise.
// ---------------------------------------------------------------------------
__global__ void bn_relu(const float* __restrict__ x, float* __restrict__ y,
                        const float* __restrict__ stats,
                        const float* __restrict__ g, const float* __restrict__ b,
                        int C, int HW, int total)
{
    int i = blockIdx.x * blockDim.x + threadIdx.x;
    if (i >= total) return;
    int c = (i / HW) % C;
    float v = (x[i] - stats[c]) * stats[C + c] * g[c] + b[c];
    y[i] = v > 0.f ? v : 0.f;
}

// ---------------------------------------------------------------------------
// AdderNet 2D, COT output channels per thread:
//   out[n,co,ho,wo] = -sum_{ci,kh,kw} |x_pad - w|
// Zero padding contributes |0 - w| = |w| (matches reference jnp.pad).
// Weights for COT channels staged into LDS via the Tensor Data Mover.
// grid.y = Cout/COT.
// ---------------------------------------------------------------------------
__global__ void adder2d_k(const float* __restrict__ x, const float* __restrict__ w,
                          float* __restrict__ out,
                          int N, int Cin, int H, int W,
                          int Cout, int K, int pad, int stride, int Ho, int Wo)
{
    extern __shared__ float wsm[]; // COT * Cin*K*K
    const int co0 = blockIdx.y * COT;
    const int nw  = Cin * K * K;
    const int tileElems = COT * nw;

#if __has_builtin(__builtin_amdgcn_tensor_load_to_lds)
    if (threadIdx.x < 32) {            // one wave issues the DMA
        tdm_load_f32_1d(w + (size_t)co0 * nw, wsm, tileElems);
        wait_tensorcnt0();
    }
#else
    for (int i = threadIdx.x; i < tileElems; i += blockDim.x)
        wsm[i] = w[(size_t)co0 * nw + i];
#endif
    __syncthreads();

    const int idx = blockIdx.x * blockDim.x + threadIdx.x;
    const int total = N * Ho * Wo;
    if (idx >= total) return;
    const int wo = idx % Wo;
    const int t  = idx / Wo;
    const int ho = t % Ho;
    const int n  = t / Ho;

    const float* xb = x + (size_t)n * Cin * H * W;
    float acc0 = 0.f, acc1 = 0.f, acc2 = 0.f, acc3 = 0.f;
    for (int ci = 0; ci < Cin; ++ci) {
        const float* xc = xb + (size_t)ci * H * W;
        const int    wb = ci * K * K;
        __builtin_prefetch(xc + ho * stride * W, 0, 1); // global_prefetch_b8
        for (int kh = 0; kh < K; ++kh) {
            const int hh   = ho * stride + kh - pad;
            const bool hin = (hh >= 0 && hh < H);
            for (int kw = 0; kw < K; ++kw) {
                const int ww = wo * stride + kw - pad;
                float xv = (hin && ww >= 0 && ww < W) ? xc[hh * W + ww] : 0.f;
                const int wi = wb + kh * K + kw;
                acc0 += fabsf(xv - wsm[0 * nw + wi]);
                acc1 += fabsf(xv - wsm[1 * nw + wi]);
                acc2 += fabsf(xv - wsm[2 * nw + wi]);
                acc3 += fabsf(xv - wsm[3 * nw + wi]);
            }
        }
    }
    const size_t cs = (size_t)Ho * Wo;
    const size_t ob = (((size_t)n * Cout + co0) * Ho + ho) * Wo + wo;
    out[ob]          = -acc0;
    out[ob + cs]     = -acc1;
    out[ob + 2 * cs] = -acc2;
    out[ob + 3 * cs] = -acc3;
}

// ---------------------------------------------------------------------------
__global__ void add_k(const float* __restrict__ a, const float* __restrict__ b,
                      float* __restrict__ o, int total)
{
    int i = blockIdx.x * blockDim.x + threadIdx.x;
    if (i < total) o[i] = a[i] + b[i];
}

__global__ void pool_k(const float* __restrict__ x, float* __restrict__ p,
                       int NC, int HW)
{
    int i = blockIdx.x * blockDim.x + threadIdx.x;
    if (i >= NC) return;
    const float* xp = x + (size_t)i * HW;
    float s = 0.f;
    for (int j = 0; j < HW; ++j) s += xp[j];
    p[i] = s / (float)HW;
}

// ---------------------------------------------------------------------------
// Host orchestration
// ---------------------------------------------------------------------------
extern "C" void kernel_launch(void* const* d_in, const int* in_sizes, int n_in,
                              void* d_out, int out_size, void* d_ws, size_t ws_size,
                              hipStream_t stream)
{
    (void)in_sizes; (void)n_in; (void)out_size; (void)ws_size;

    const float* x       = (const float*)d_in[0];
    const float* conv1_w = (const float*)d_in[1];
    const float* conv1_b = (const float*)d_in[2];

    struct Blk { const float *bn1_g,*bn1_b,*w1,*bn2_g,*bn2_b,*w2,*wsc;
                 int cin, cout, stride; };
    const int BL[6][3] = {{16,64,1},{64,64,1},{64,128,2},
                          {128,128,1},{128,256,2},{256,256,1}};
    Blk blk[6];
    int idx = 3;
    for (int i = 0; i < 6; ++i) {
        blk[i].cin = BL[i][0]; blk[i].cout = BL[i][1]; blk[i].stride = BL[i][2];
        blk[i].bn1_g = (const float*)d_in[idx++];
        blk[i].bn1_b = (const float*)d_in[idx++];
        blk[i].w1    = (const float*)d_in[idx++];
        blk[i].bn2_g = (const float*)d_in[idx++];
        blk[i].bn2_b = (const float*)d_in[idx++];
        blk[i].w2    = (const float*)d_in[idx++];
        blk[i].wsc   = (blk[i].stride != 1 || blk[i].cin != blk[i].cout)
                         ? (const float*)d_in[idx++] : nullptr;
    }
    const float* bnf_g = (const float*)d_in[idx++];
    const float* bnf_b = (const float*)d_in[idx++];
    const float* fc_w  = (const float*)d_in[idx++];
    const float* fc_b  = (const float*)d_in[idx++];

    float* ws = (float*)d_ws;
    const size_t BUF = 2u * 1024u * 1024u; // elements; max activation = 2M floats
    float* F[4]   = { ws, ws + BUF, ws + 2 * BUF, ws + 3 * BUF };
    float* stats  = ws + 4 * BUF;     // 512 floats
    float* pooled = stats + 512;      // 16*256 floats

    const int Nb = 16;

    conv1_wmma<<<128, 256, 0, stream>>>(x, conv1_w, conv1_b, F[0]);

    int cur = 0, H = 32, W = 32;
    for (int i = 0; i < 6; ++i) {
        const int cin = blk[i].cin, cout = blk[i].cout, stride = blk[i].stride;
        const int HW = H * W;
        const int a = (cur + 1) & 3, bb = (cur + 2) & 3, cc = (cur + 3) & 3;

        // h = relu(bn1(out))
        bn_stats<<<cin, 256, 0, stream>>>(F[cur], stats, cin, HW, Nb);
        int tot1 = Nb * cin * HW;
        bn_relu<<<(tot1 + 255) / 256, 256, 0, stream>>>(
            F[cur], F[a], stats, blk[i].bn1_g, blk[i].bn1_b, cin, HW, tot1);

        // h = adder2d(h, w1, stride=1, pad=1)
        {
            dim3 g((Nb * H * W + 255) / 256, cout / COT);
            adder2d_k<<<g, 256, COT * cin * 9 * sizeof(float), stream>>>(
                F[a], blk[i].w1, F[bb], Nb, cin, H, W, cout, 3, 1, 1, H, W);
        }

        // h = relu(bn2(h))  (in place)
        bn_stats<<<cout, 256, 0, stream>>>(F[bb], stats, cout, HW, Nb);
        int tot2 = Nb * cout * HW;
        bn_relu<<<(tot2 + 255) / 256, 256, 0, stream>>>(
            F[bb], F[bb], stats, blk[i].bn2_g, blk[i].bn2_b, cout, HW, tot2);

        // h = adder2d(h, w2, stride, pad=1)
        const int Ho = (H - 1) / stride + 1, Wo = (W - 1) / stride + 1;
        {
            dim3 g((Nb * Ho * Wo + 255) / 256, cout / COT);
            adder2d_k<<<g, 256, COT * cout * 9 * sizeof(float), stream>>>(
                F[bb], blk[i].w2, F[cc], Nb, cout, H, W, cout, 3, 1, stride, Ho, Wo);
        }

        // shortcut
        const float* sc;
        if (blk[i].wsc) {
            dim3 g((Nb * Ho * Wo + 255) / 256, cout / COT);
            adder2d_k<<<g, 256, COT * cin * sizeof(float), stream>>>(
                F[cur], blk[i].wsc, F[a], Nb, cin, H, W, cout, 1, 0, stride, Ho, Wo);
            sc = F[a];
        } else {
            sc = F[cur];
        }

        // out = h + sc
        int tot3 = Nb * cout * Ho * Wo;
        add_k<<<(tot3 + 255) / 256, 256, 0, stream>>>(F[cc], sc, F[a], tot3);

        cur = a; H = Ho; W = Wo;
    }

    // final: relu(bn(out)) -> global avg pool -> fc
    const int C = 256, HW = H * W;
    bn_stats<<<C, 256, 0, stream>>>(F[cur], stats, C, HW, Nb);
    int tot = Nb * C * HW;
    bn_relu<<<(tot + 255) / 256, 256, 0, stream>>>(
        F[cur], F[cur], stats, bnf_g, bnf_b, C, HW, tot);
    pool_k<<<(Nb * C + 255) / 256, 256, 0, stream>>>(F[cur], pooled, Nb * C, HW);
    fc_wmma<<<1, 32, 0, stream>>>(pooled, fc_w, fc_b, (float*)d_out);
}
